// ResGATsAblation_59906203845003
// MI455X (gfx1250) — matrix operand order repaired and checked by reading, not compiled, not run on hardware
//
#include <hip/hip_runtime.h>
#include <math.h>

#define NN 10000
#define NG 16

typedef __attribute__((ext_vector_type(16))) __bf16   v16bf;
typedef __attribute__((ext_vector_type(8)))  float    v8f;
typedef __attribute__((ext_vector_type(4)))  unsigned v4u;
typedef __attribute__((ext_vector_type(8)))  unsigned v8u;

__device__ __forceinline__ unsigned short f2bf(float f) {
    unsigned u = __float_as_uint(f);
    u = u + 0x7fffu + ((u >> 16) & 1u);
    return (unsigned short)(u >> 16);
}
__device__ __forceinline__ float eluf(float x)   { return x > 0.f ? x : expm1f(x); }
__device__ __forceinline__ float lrelu02(float x){ return x > 0.f ? x : 0.2f * x; }

__device__ __forceinline__ void atomicMaxF(float* a, float v) {
    unsigned* ua = (unsigned*)a;
    unsigned cur = *ua;
    while (v > __uint_as_float(cur)) {
        unsigned prev = atomicCAS(ua, cur, __float_as_uint(v));
        if (prev == cur) break;
        cur = prev;
    }
}

// ---------------- utility kernels ----------------
__global__ void k_zero(float* p, int n) {
    int i = blockIdx.x * blockDim.x + threadIdx.x;
    if (i < n) p[i] = 0.f;
}
__global__ void k_fill(float* p, float v, int n) {
    int i = blockIdx.x * blockDim.x + threadIdx.x;
    if (i < n) p[i] = v;
}
__global__ void k_counts(const int* batch, float* cnt, int n) {
    int i = blockIdx.x * blockDim.x + threadIdx.x;
    if (i < n) atomicAdd(&cnt[batch[i]], 1.0f);
}
__global__ void k_f2bf(const float* s, unsigned short* d, int n) {
    int i = blockIdx.x * blockDim.x + threadIdx.x;
    if (i < n) d[i] = f2bf(s[i]);
}
// W (K x Nc, row-major f32) -> Wt (Nc x K, bf16)
__global__ void k_wt_bf(const float* W, unsigned short* Wt, int K, int Nc) {
    int i = blockIdx.x * blockDim.x + threadIdx.x;
    if (i >= K * Nc) return;
    int k = i / Nc, n = i - k * Nc;
    Wt[(size_t)n * K + k] = f2bf(W[i]);
}

// ---------------- WMMA GEMM: C = A(bf16 MxK) @ B(Wt: Nc x K bf16) + bias --------------
// 128 threads (4 waves). Workgroup tile: 64(M) x 64(N). Wave w owns rows
// [mtile*64 + w*16, +16) and all 64 columns as 4 accumulators. The shared
// 64-col x 32-k B panel is staged to LDS per K-step with async copies.
__global__ void k_gemm_wmma(const unsigned short* __restrict__ A,
                            const unsigned short* __restrict__ Bt,
                            const float* __restrict__ bias,
                            float* __restrict__ C,
                            int M, int K, int Nc) {
    __shared__ unsigned short Bsh[64 * 32];   // 64 cols x 32 k bf16 = 4 KB

    int tid  = threadIdx.x;                   // 0..127
    int wave = tid >> 5;
    int lane = tid & 31;
    int l16  = lane & 15;
    int half = lane >> 4;
    int colbase = blockIdx.x * 64;
    int mrow = blockIdx.y * 64 + wave * 16 + l16;
    int mclamp = mrow < M ? mrow : (M - 1);

    v8f acc[4];
#pragma unroll
    for (int j = 0; j < 4; ++j) {
        float bv = bias[colbase + j * 16 + l16];
#pragma unroll
        for (int r = 0; r < 8; ++r) acc[j][r] = bv;
    }

    const unsigned short* arow = A + (size_t)mclamp * K;

    // async-copy assignment: thread t moves 2x16B of column (t>>1), chunk (t&1)
    int cc = tid >> 1, q = tid & 1;
    const unsigned short* gB0 = Bt + (size_t)(colbase + cc) * K + q * 8;
    unsigned ldsB = ((unsigned)(unsigned long long)(void*)Bsh) + cc * 64 + q * 16;

    for (int k0 = 0; k0 < K; k0 += 32) {
        unsigned long long gp = (unsigned long long)(gB0 + k0);
        asm volatile("global_load_async_to_lds_b128 %0, %1, off"
                     :: "v"(ldsB), "v"(gp) : "memory");
        asm volatile("global_load_async_to_lds_b128 %0, %1, off offset:32"
                     :: "v"(ldsB), "v"(gp) : "memory");
        asm volatile("s_wait_asynccnt 0" ::: "memory");
        __syncthreads();

        int ab = k0 + half * 16;
        v4u lo = *(const v4u*)(arow + ab);
        v4u hi = *(const v4u*)(arow + ab + 8);
        v8u au = __builtin_shufflevector(lo, hi, 0, 1, 2, 3, 4, 5, 6, 7);
        v16bf a = __builtin_bit_cast(v16bf, au);
#pragma unroll
        for (int j = 0; j < 4; ++j) {
            v16bf b = *(const v16bf*)(Bsh + (j * 16 + l16) * 32 + half * 16);
            acc[j] = __builtin_amdgcn_wmma_f32_16x16x32_bf16(false, a, false, b,
                                                             (short)0, acc[j], false, false);
        }
        __syncthreads();   // all waves done reading before next overwrite
    }

    int mb = blockIdx.y * 64 + wave * 16 + half * 8;
#pragma unroll
    for (int j = 0; j < 4; ++j) {
        int col = colbase + j * 16 + l16;
#pragma unroll
        for (int r = 0; r < 8; ++r) {
            int m = mb + r;
            if (m < M) C[(size_t)m * Nc + col] = acc[j][r];
        }
    }
}

// ---------------- GATv2 attention kernels ----------------
__global__ void k_init_out(float* OUT, const float* cvb, int D, int total) {
    int i = blockIdx.x * blockDim.x + threadIdx.x;
    if (i < total) OUT[i] = cvb[i % D];
}
__global__ void k_edge_logits(const int* srcA, const int* dstA,
                              const float* XL, const float* XR, const float* att,
                              float* elog, float* mmax,
                              int E, int E2, int H, int C, int D) {
    int i = blockIdx.x * blockDim.x + threadIdx.x;
    if (i >= E2 * H) return;
    int e = i / H, h = i - e * H;
    int s = e < E ? srcA[e] : (e - E);
    int d = e < E ? dstA[e] : (e - E);
    const float* xl = XL + (size_t)s * D + h * C;
    const float* xr = XR + (size_t)d * D + h * C;
    const float* at = att + h * C;
    float acc = 0.f;
    for (int c = 0; c < C; ++c) acc += at[c] * lrelu02(xl[c] + xr[c]);
    elog[i] = acc;
    atomicMaxF(&mmax[d * H + h], acc);
}
__global__ void k_edge_exp(const int* dstA, const float* elog, const float* mmax,
                           float* eexp, float* ssum, int E, int E2, int H) {
    int i = blockIdx.x * blockDim.x + threadIdx.x;
    if (i >= E2 * H) return;
    int e = i / H, h = i - e * H;
    int d = e < E ? dstA[e] : (e - E);
    float ex = expf(elog[i] - mmax[d * H + h]);
    eexp[i] = ex;
    atomicAdd(&ssum[d * H + h], ex);
}
__global__ void k_edge_scatter(const int* srcA, const int* dstA,
                               const float* XL, const float* eexp, const float* ssum,
                               float* OUT, int E, int E2, int H, int C, int D) {
    int i = blockIdx.x * blockDim.x + threadIdx.x;
    if (i >= E2 * D) return;
    int e = i / D, dd = i - e * D;
    int h = dd / C;
    int s = e < E ? srcA[e] : (e - E);
    int d = e < E ? dstA[e] : (e - E);
    float alpha = eexp[e * H + h] / (ssum[d * H + h] + 1e-16f);
    atomicAdd(&OUT[(size_t)d * D + dd], alpha * XL[(size_t)s * D + dd]);
}

// ---------------- GraphNorm ----------------
__global__ void k_seg_sum(const float* X, const int* batch, float* acc, int D, int total) {
    int i = blockIdx.x * blockDim.x + threadIdx.x;
    if (i >= total) return;
    int n = i / D, d = i - n * D;
    atomicAdd(&acc[batch[n] * D + d], X[i]);
}
__global__ void k_center(float* X, const int* batch, const float* accMean,
                         const float* aP, const float* cnt, float* accVar,
                         int D, int total) {
    int i = blockIdx.x * blockDim.x + threadIdx.x;
    if (i >= total) return;
    int n = i / D, d = i - n * D;
    int g = batch[n];
    float c = fmaxf(cnt[g], 1.0f);
    float v = X[i] - aP[d] * (accMean[g * D + d] / c);
    X[i] = v;
    atomicAdd(&accVar[g * D + d], v * v);
}
__global__ void k_gn_apply(float* X, const int* batch, const float* accVar,
                           const float* gP, const float* bP, const float* cnt,
                           int D, int total) {
    int i = blockIdx.x * blockDim.x + threadIdx.x;
    if (i >= total) return;
    int n = i / D, d = i - n * D;
    int g = batch[n];
    float c = fmaxf(cnt[g], 1.0f);
    X[i] = gP[d] * X[i] * rsqrtf(accVar[g * D + d] / c + 1e-5f) + bP[d];
}
__global__ void k_combine(float* OUT, const float* SK, unsigned short* actbf, int total) {
    int i = blockIdx.x * blockDim.x + threadIdx.x;
    if (i >= total) return;
    float v = eluf(OUT[i] + SK[i]);
    OUT[i] = v;
    actbf[i] = f2bf(v);
}

// ---------------- head ----------------
__global__ void k_pool(const float* Hf, const int* batch, float* pool, int D, int total) {
    int i = blockIdx.x * blockDim.x + threadIdx.x;
    if (i >= total) return;
    int n = i / D, d = i - n * D;
    atomicAdd(&pool[batch[n] * D + d], Hf[i]);
}
__global__ void k_fc1(const float* pool, const float* W, const float* b,
                      const float* cnt, float* out) {
    int i = blockIdx.x * blockDim.x + threadIdx.x;
    if (i >= NG * 512) return;
    int g = i / 512, j = i - g * 512;
    float inv = 1.0f / fmaxf(cnt[g], 1.0f);
    float s = b[j];
    for (int d = 0; d < 256; ++d) s += pool[g * 256 + d] * inv * W[d * 512 + j];
    out[i] = eluf(s);
}
__global__ void k_fc2(const float* h, const float* W, const float* b, float* out) {
    int g = threadIdx.x;
    if (g >= NG) return;
    float l0 = b[0], l1 = b[1];
    for (int j = 0; j < 512; ++j) {
        float v = h[g * 512 + j];
        l0 += v * W[j * 2 + 0];
        l1 += v * W[j * 2 + 1];
    }
    out[g * 2 + 0] = l0;
    out[g * 2 + 1] = l1;
    float m = fmaxf(l0, l1);
    float lse = m + logf(expf(l0 - m) + expf(l1 - m));
    out[32 + g * 2 + 0] = l0 - lse;
    out[32 + g * 2 + 1] = l1 - lse;
}

extern "C" void kernel_launch(void* const* d_in, const int* in_sizes, int n_in,
                              void* d_out, int out_size, void* d_ws, size_t ws_size,
                              hipStream_t stream) {
    const float* x   = (const float*)d_in[0];
    const int* eidx  = (const int*)d_in[1];
    const int* batch = (const int*)d_in[2];
    const int E  = in_sizes[1] / 2;
    const int E2 = E + NN;
    const int* srcA = eidx;
    const int* dstA = eidx + E;

    const float* P[64];
    for (int i = 0; i < n_in && i < 64; ++i) P[i] = (const float*)d_in[i];

    const int Din[3]  = {128, 1024, 512};
    const int DoutA[3]= {1024, 512, 256};
    const int Hh[3]   = {8, 2, 4};
    const int base[3] = {3, 17, 31};

    char* w = (char*)d_ws;
    size_t off = 0;
    auto alloc = [&](size_t bytes) -> void* {
        off = (off + 255) & ~(size_t)255;
        void* p = w + off;
        off += bytes;
        return p;
    };
    unsigned short* ACT = (unsigned short*)alloc((size_t)NN * 1024 * 2);
    float* XL  = (float*)alloc((size_t)NN * 1024 * 4);
    float* XS  = (float*)alloc((size_t)NN * 1024 * 4);
    float* OUT = (float*)alloc((size_t)NN * 1024 * 4);
    unsigned short* WT[3];
    for (int i = 0; i < 3; ++i) WT[i] = (unsigned short*)alloc((size_t)1024 * 512 * 2);
    float* ELOG = (float*)alloc((size_t)E2 * 8 * 4);
    float* EEXP = (float*)alloc((size_t)E2 * 8 * 4);
    float* MMAX = (float*)alloc((size_t)NN * 8 * 4);
    float* SSUM = (float*)alloc((size_t)NN * 8 * 4);
    float* GA   = (float*)alloc((size_t)NG * 1024 * 4);
    float* GB   = (float*)alloc((size_t)NG * 1024 * 4);
    float* CNT  = (float*)alloc(256);
    float* POOL = (float*)alloc((size_t)NG * 256 * 4);
    float* F1   = (float*)alloc((size_t)NG * 512 * 4);
    (void)ws_size; (void)out_size;

    auto nb = [](long long n) { return (unsigned)((n + 255) / 256); };

    k_zero<<<1, 32, 0, stream>>>(CNT, NG);
    k_counts<<<nb(NN), 256, 0, stream>>>(batch, CNT, NN);
    k_f2bf<<<nb((long long)NN * 128), 256, 0, stream>>>(x, ACT, NN * 128);

    for (int b = 0; b < 3; ++b) {
        int K = Din[b], D = DoutA[b], H = Hh[b], C = D / H;
        const float* Wl  = P[base[b] + 0];  const float* bl  = P[base[b] + 1];
        const float* Wr  = P[base[b] + 2];  const float* br  = P[base[b] + 3];
        const float* att = P[base[b] + 4];  const float* cvb = P[base[b] + 5];
        const float* skW = P[base[b] + 6];  const float* skb = P[base[b] + 7];
        const float* gng = P[base[b] + 8];  const float* gnb = P[base[b] + 9];
        const float* gna = P[base[b] + 10];
        const float* sng = P[base[b] + 11]; const float* snb = P[base[b] + 12];
        const float* sna = P[base[b] + 13];

        long long KD = (long long)K * D;
        k_wt_bf<<<nb(KD), 256, 0, stream>>>(Wl,  WT[0], K, D);
        k_wt_bf<<<nb(KD), 256, 0, stream>>>(Wr,  WT[1], K, D);
        k_wt_bf<<<nb(KD), 256, 0, stream>>>(skW, WT[2], K, D);

        dim3 gg((D + 63) / 64, (NN + 63) / 64);
        k_gemm_wmma<<<gg, 128, 0, stream>>>(ACT, WT[0], bl, XL, NN, K, D);
        k_gemm_wmma<<<gg, 128, 0, stream>>>(ACT, WT[1], br, XS, NN, K, D);

        k_fill<<<nb(NN * H), 256, 0, stream>>>(MMAX, -1e30f, NN * H);
        k_zero<<<nb(NN * H), 256, 0, stream>>>(SSUM, NN * H);
        long long ND = (long long)NN * D;
        k_init_out<<<nb(ND), 256, 0, stream>>>(OUT, cvb, D, (int)ND);

        k_edge_logits<<<nb((long long)E2 * H), 256, 0, stream>>>(
            srcA, dstA, XL, XS, att, ELOG, MMAX, E, E2, H, C, D);
        k_edge_exp<<<nb((long long)E2 * H), 256, 0, stream>>>(
            dstA, ELOG, MMAX, EEXP, SSUM, E, E2, H);
        k_edge_scatter<<<nb((long long)E2 * D), 256, 0, stream>>>(
            srcA, dstA, XL, EEXP, SSUM, OUT, E, E2, H, C, D);

        // skip-connection GEMM reuses XS (XR no longer needed)
        k_gemm_wmma<<<gg, 128, 0, stream>>>(ACT, WT[2], skb, XS, NN, K, D);

        // GraphNorm on aggregated output (gn params)
        k_zero<<<nb(NG * D), 256, 0, stream>>>(GA, NG * D);
        k_zero<<<nb(NG * D), 256, 0, stream>>>(GB, NG * D);
        k_seg_sum<<<nb(ND), 256, 0, stream>>>(OUT, batch, GA, D, (int)ND);
        k_center<<<nb(ND), 256, 0, stream>>>(OUT, batch, GA, gna, CNT, GB, D, (int)ND);
        k_gn_apply<<<nb(ND), 256, 0, stream>>>(OUT, batch, GB, gng, gnb, CNT, D, (int)ND);

        // GraphNorm on skip (sn params)
        k_zero<<<nb(NG * D), 256, 0, stream>>>(GA, NG * D);
        k_zero<<<nb(NG * D), 256, 0, stream>>>(GB, NG * D);
        k_seg_sum<<<nb(ND), 256, 0, stream>>>(XS, batch, GA, D, (int)ND);
        k_center<<<nb(ND), 256, 0, stream>>>(XS, batch, GA, sna, CNT, GB, D, (int)ND);
        k_gn_apply<<<nb(ND), 256, 0, stream>>>(XS, batch, GB, sng, snb, CNT, D, (int)ND);

        k_combine<<<nb(ND), 256, 0, stream>>>(OUT, XS, ACT, (int)ND);
    }

    k_zero<<<nb(NG * 256), 256, 0, stream>>>(POOL, NG * 256);
    k_pool<<<nb((long long)NN * 256), 256, 0, stream>>>(OUT, batch, POOL, 256, NN * 256);
    k_fc1<<<nb(NG * 512), 256, 0, stream>>>(POOL, P[45], P[46], CNT, F1);
    k_fc2<<<1, 32, 0, stream>>>(F1, P[47], P[48], (float*)d_out);
}
